// qMT_SPGR_15753940042254
// MI455X (gfx1250) — compile-verified, matrix operand order checked by reading, not run on hardware
//
#include <hip/hip_runtime.h>

typedef __attribute__((ext_vector_type(16))) _Float16 v16h;
typedef __attribute__((ext_vector_type(8)))  float    v8f;

#define PI_F   3.14159265358979323846f
#define LOG2E  1.4426950408889634f

// d_ws float layout:
// [0]=cos(exc) [1]=sin(exc) [2]=Wfac=pi*w1^2*mt_dur/tr [3]=tr
// [16..143]   Acoef[128]  (sqrt(2/pi)*trapw*sin(theta)/term, zero padded)
// [160..287]  Bcoef[128]  (2*(2*pi*delta)^2*log2(e)/term^2, zero padded)

__global__ void qmt_setup(const float* __restrict__ tr_p,
                          const float* __restrict__ exc_p,
                          const float* __restrict__ mtfa_p,
                          const float* __restrict__ moff_p,
                          const float* __restrict__ mdur_p,
                          float* __restrict__ ws) {
  const int i = threadIdx.x;
  const float tr    = tr_p[0];
  const float mdur  = mdur_p[0];
  const float delta = moff_p[0];
  if (i < 128) {
    float A = 0.0f, Bc = 0.0f;
    if (i < 100) {
      const float dth = (PI_F * 0.5f) / 99.0f;
      const float th  = dth * (float)i;
      const float st  = sinf(th);
      const float ct  = cosf(th);
      float term = fabsf(3.0f * ct * ct - 1.0f);
      term = fmaxf(term, 1e-6f);
      const float inv = 1.0f / term;
      const float tw  = dth * ((i == 0 || i == 99) ? 0.5f : 1.0f);
      A = 0.79788456080286536f * tw * st * inv;        // sqrt(2/pi)
      const float w = 2.0f * PI_F * delta;
      const float Ck = 2.0f * w * w * LOG2E;           // exp(x) -> exp2(x*log2e)
      Bc = Ck * inv * inv;
    }
    ws[16 + i]  = A;
    ws[160 + i] = Bc;
  }
  if (i == 0) {
    const float exc_rad = exc_p[0]  * (PI_F / 180.0f);
    const float mt_rad  = mtfa_p[0] * (PI_F / 180.0f);
    const float w1 = mt_rad / mdur;
    ws[0] = cosf(exc_rad);
    ws[1] = sinf(exc_rad);
    ws[2] = PI_F * w1 * w1 * (mdur / tr);
    ws[3] = tr;
  }
}

__device__ __forceinline__ v8f wmma_acc(v16h a, v16h b1, v8f c) {
  return __builtin_amdgcn_wmma_f32_16x16x32_f16(false, a, false, b1,
                                                (short)0, c, false, false);
}

__launch_bounds__(256)
__global__ void qmt_main(const float* __restrict__ f_,
                         const float* __restrict__ kmf_,
                         const float* __restrict__ r1f_,
                         const float* __restrict__ r1m_,
                         const float* __restrict__ t2m_,
                         const float* __restrict__ ws,
                         float* __restrict__ out, int n) {
  __shared__ __align__(16) float sAc[128];
  __shared__ __align__(16) float sBc[128];
  __shared__ float sHdr[4];
  __shared__ float sSum[8 * 32];

  const int tid = threadIdx.x;
  if (tid < 128) { sAc[tid] = ws[16 + tid]; sBc[tid] = ws[160 + tid]; }
  if (tid < 4)   { sHdr[tid] = ws[tid]; }
  __syncthreads();

  const int wv   = tid >> 5;
  const int lane = tid & 31;
  const int hf   = lane >> 4;     // K-half of the wave (A-matrix layout)
  const int lm   = lane & 15;
  const int vb   = (blockIdx.x * 8 + wv) * 32;

  // per-lane T2m^2 for both 16-voxel groups this wave reduces via WMMA
  int ia = vb + lm;       if (ia >= n) ia = n - 1;
  int ib = vb + 16 + lm;  if (ib >= n) ib = n - 1;
  float t2a = t2m_[ia]; t2a *= t2a;
  float t2b = t2m_[ib]; t2b *= t2b;

  v16h bOne;
#pragma unroll
  for (int j = 0; j < 16; ++j) bOne[j] = (_Float16)1.0f;

  v8f acc0 = {}; // voxels vb .. vb+15
  v8f acc1 = {}; // voxels vb+16 .. vb+31

  // Full chunks: K = 32c .. 32c+31.  Lane holds K = klo..klo+7, klo+16..klo+23.
#pragma unroll
  for (int c = 0; c < 3; ++c) {
    const int klo = (c << 5) + (hf << 3);
    const float4 b0 = *(const float4*)(sBc + klo);
    const float4 b1 = *(const float4*)(sBc + klo + 4);
    const float4 b2 = *(const float4*)(sBc + klo + 16);
    const float4 b3 = *(const float4*)(sBc + klo + 20);
    const float4 w0 = *(const float4*)(sAc + klo);
    const float4 w1 = *(const float4*)(sAc + klo + 4);
    const float4 w2 = *(const float4*)(sAc + klo + 16);
    const float4 w3 = *(const float4*)(sAc + klo + 20);
    const float Bv[16] = {b0.x,b0.y,b0.z,b0.w, b1.x,b1.y,b1.z,b1.w,
                          b2.x,b2.y,b2.z,b2.w, b3.x,b3.y,b3.z,b3.w};
    const float Wv[16] = {w0.x,w0.y,w0.z,w0.w, w1.x,w1.y,w1.z,w1.w,
                          w2.x,w2.y,w2.z,w2.w, w3.x,w3.y,w3.z,w3.w};
    v16h aA, aB;
#pragma unroll
    for (int j = 0; j < 16; ++j) {
      aA[j] = (_Float16)(Wv[j] * __builtin_amdgcn_exp2f(-t2a * Bv[j]));
      aB[j] = (_Float16)(Wv[j] * __builtin_amdgcn_exp2f(-t2b * Bv[j]));
    }
    acc0 = wmma_acc(aA, bOne, acc0);
    acc1 = wmma_acc(aB, bOne, acc1);
  }

  // Tail chunk: only K=96..99 are real; they live in half 0, elements 0..3.
  {
    v16h aA = {}; v16h aB = {};
    const float4 b4 = *(const float4*)(sBc + 96);
    const float4 w4 = *(const float4*)(sAc + 96);
    if (hf == 0) {
      aA[0] = (_Float16)(w4.x * __builtin_amdgcn_exp2f(-t2a * b4.x));
      aA[1] = (_Float16)(w4.y * __builtin_amdgcn_exp2f(-t2a * b4.y));
      aA[2] = (_Float16)(w4.z * __builtin_amdgcn_exp2f(-t2a * b4.z));
      aA[3] = (_Float16)(w4.w * __builtin_amdgcn_exp2f(-t2a * b4.w));
      aB[0] = (_Float16)(w4.x * __builtin_amdgcn_exp2f(-t2b * b4.x));
      aB[1] = (_Float16)(w4.y * __builtin_amdgcn_exp2f(-t2b * b4.y));
      aB[2] = (_Float16)(w4.z * __builtin_amdgcn_exp2f(-t2b * b4.z));
      aB[3] = (_Float16)(w4.w * __builtin_amdgcn_exp2f(-t2b * b4.w));
    }
    acc0 = wmma_acc(aA, bOne, acc0);  // EXEC is all-1s again here
    acc1 = wmma_acc(aB, bOne, acc1);
  }

  // All columns of D are identical (B = ones): column 0 lanes export the sums.
  // C/D layout: lane 0 -> M=0..7 in vgpr 0..7, lane 16 -> M=8..15.
  if (lm == 0) {
#pragma unroll
    for (int r = 0; r < 8; ++r) {
      sSum[wv * 32 + (hf << 3) + r]      = acc0[r];
      sSum[wv * 32 + 16 + (hf << 3) + r] = acc1[r];
    }
  }
  __syncthreads();

  // ---- per-lane closed-form 2x2 Bloch-McConnell tail (one voxel per lane) ----
  const int v  = vb + lane;
  const int vi = (v < n) ? v : n - 1;
  const float S    = sSum[wv * 32 + lane];
  const float f    = f_[vi];
  const float kmf  = kmf_[vi];
  const float R1f  = r1f_[vi];
  const float R1m  = r1m_[vi];
  const float T2m  = t2m_[vi];
  const float ca   = sHdr[0];
  const float sa   = sHdr[1];
  const float Wfac = sHdr[2];
  const float tr   = sHdr[3];

  const float G   = T2m * S;               // super-Lorentzian integral
  const float W   = Wfac * G;              // saturation rate
  const float kfr = kmf * f / (1.0f - f + 1e-9f);

  // R = [[a,b],[c,d]]
  const float a = -(R1f + kfr);
  const float b = kmf;
  const float c = kfr;
  const float d = -(R1m + kmf + W);

  // E = expm(tr*R), closed form (real eigenvalues: b,c > 0)
  const float ma = a * tr, mb = b * tr, mc = c * tr, md = d * tr;
  const float mu = 0.5f * (ma + md);
  const float p  = 0.5f * (ma - md);
  float q2 = p * p + mb * mc;
  q2 = fmaxf(q2, 0.0f);
  const float q   = sqrtf(q2);
  const float eq  = __builtin_amdgcn_exp2f(q * LOG2E);
  const float eiq = 1.0f / eq;
  const float ch  = 0.5f * (eq + eiq);
  const float sh  = 0.5f * (eq - eiq);
  const float shq = (q > 1e-4f) ? (sh / q) : (1.0f + 0.16666667f * q2);
  const float emu = __builtin_amdgcn_exp2f(mu * LOG2E);
  const float E00 = emu * (ch + p * shq);
  const float E01 = emu * (mb * shq);
  const float E10 = emu * (mc * shq);
  const float E11 = emu * (ch - p * shq);

  // M_relax = R^-1 (E - I) B
  const float B0 = R1f * (1.0f - f);
  const float B1 = R1m * f;
  const float v0 = (E00 - 1.0f) * B0 + E01 * B1;
  const float v1 = E10 * B0 + (E11 - 1.0f) * B1;
  const float rdet = 1.0f / (a * d - b * c);
  const float mr0  = ( d * v0 - b * v1) * rdet;
  const float mr1  = (-c * v0 + a * v1) * rdet;

  // solve (I - E*diag(ca,1)) x = M_relax ; out = x0 * sa
  const float M00 = 1.0f - E00 * ca;
  const float M01 = -E01;
  const float M10 = -E10 * ca;
  const float M11 = 1.0f - E11;
  const float x0  = (mr0 * M11 - M01 * mr1) / (M00 * M11 - M01 * M10);

  if (v < n) out[v] = x0 * sa;
}

extern "C" void kernel_launch(void* const* d_in, const int* in_sizes, int n_in,
                              void* d_out, int out_size, void* d_ws, size_t ws_size,
                              hipStream_t stream) {
  const float* f   = (const float*)d_in[0];
  const float* kmf = (const float*)d_in[1];
  const float* r1f = (const float*)d_in[2];
  const float* r1m = (const float*)d_in[3];
  // d_in[4] = T2_f (unused by the model)
  const float* t2m = (const float*)d_in[5];
  const float* trp = (const float*)d_in[6];
  const float* exc = (const float*)d_in[7];
  const float* mtf = (const float*)d_in[8];
  const float* mof = (const float*)d_in[9];
  const float* mdu = (const float*)d_in[10];
  float* out = (float*)d_out;
  float* ws  = (float*)d_ws;
  const int n = in_sizes[0];

  qmt_setup<<<1, 128, 0, stream>>>(trp, exc, mtf, mof, mdu, ws);

  const int vox_per_block = 256;            // 8 waves * 32 voxels
  const int blocks = (n + vox_per_block - 1) / vox_per_block;
  qmt_main<<<blocks, 256, 0, stream>>>(f, kmf, r1f, r1m, t2m, ws, out, n);
}
